// SelfAttentionBlock_33646773797395
// MI455X (gfx1250) — matrix operand order repaired and checked by reading, not compiled, not run on hardware
//
#include <hip/hip_runtime.h>

// ---------------------------------------------------------------------------
// SelfAttentionBlock for MI455X (gfx1250), bf16 WMMA, flash-attention fusion.
// B=2, C=512, H=W=64 (N=4096), heads=4, head_dim=64, scale=0.125
// Round 3: fix async builtin pointer types (V4i AS1 src, V4i AS3 dst).
// ---------------------------------------------------------------------------

typedef __attribute__((ext_vector_type(16))) __bf16 v16bf;
typedef __attribute__((ext_vector_type(8)))  __bf16 v8bf;
typedef __attribute__((ext_vector_type(4)))  __bf16 v4bf;
typedef __attribute__((ext_vector_type(8)))  float  v8f;
typedef __attribute__((ext_vector_type(4)))  float  v4f;
typedef __attribute__((ext_vector_type(4)))  unsigned int v4u;
typedef __attribute__((ext_vector_type(8)))  int v8i_t;
typedef __attribute__((ext_vector_type(4)))  int v4i_t;

#define N_TOK 4096
#define C_IN  512
#define HEADS 4
#define HDIM  64
#define SCALE 0.125f

#if defined(__HIP_DEVICE_COMPILE__) && defined(__gfx1250__)
  #if __has_builtin(__builtin_amdgcn_tensor_load_to_lds)
    #define HAVE_TDM 1
  #endif
  #if __has_builtin(__builtin_amdgcn_global_load_async_to_lds_b128)
    #define HAVE_ASYNC 1
  #endif
#endif

__device__ __forceinline__ v8f wmma_bf16(v16bf a, v16bf b, v8f c) {
  return __builtin_amdgcn_wmma_f32_16x16x32_bf16(
      /*neg_a=*/false, a, /*neg_b=*/false, b,
      /*c_mod=*/(short)0, c, /*reuse_a=*/false, /*reuse_b=*/false);
}

// A fragment: 16x32 (MxK) bf16 from row-major LDS tile (ISA 7.12.2 layout).
__device__ __forceinline__ v16bf load_A(const __bf16* base, int pitch, int m0, int k0) {
  const int lane = threadIdx.x & 31;
  const int half = lane >> 4;
  const __bf16* row = base + (m0 + (lane & 15)) * pitch + k0 + half * 8;
  v16bf a;
#pragma unroll
  for (int e = 0; e < 8; ++e) { a[e] = row[e]; a[e + 8] = row[e + 16]; }
  return a;
}

// B fragment: 32x16 (KxN) bf16; lane L holds column N=n0+L%16, K in table order.
__device__ __forceinline__ v16bf load_B(const __bf16* base, int pitch, int k0, int n0) {
  const int lane = threadIdx.x & 31;
  const int half = lane >> 4;
  const __bf16* col = base + (k0 + half * 8) * pitch + n0 + (lane & 15);
  v16bf b;
#pragma unroll
  for (int e = 0; e < 8; ++e) { b[e] = col[e * pitch]; b[e + 8] = col[(e + 16) * pitch]; }
  return b;
}

// f32x4 global load -> bf16x4 LDS store (packed cvt).
__device__ __forceinline__ void cvt_store4(__bf16* dst, const float* src) {
  const v4f f = *(const v4f*)src;
  v4bf h;
#pragma unroll
  for (int e = 0; e < 4; ++e) h[e] = (__bf16)f[e];
  *(v4bf*)dst = h;
}

#if defined(HAVE_ASYNC)
typedef __attribute__((address_space(1))) v4i_t as1_v4i;
typedef __attribute__((address_space(3))) v4i_t as3_v4i;

__device__ __forceinline__ as1_v4i* to_as1(const void* p) {
  return (as1_v4i*)(uintptr_t)p;
}
__device__ __forceinline__ as3_v4i* to_as3(void* p) {
  return (as3_v4i*)(uintptr_t)p;
}
#endif

#if defined(HAVE_TDM)
// One TDM descriptor: 2D tile (tile_d1 rows x tile_d0 elems) of 2-byte data
// out of a (tensor_d1 x tensor_d0) row-major tensor, into LDS with
// 8-bf16 (4-dword) padding every 32 dwords -> LDS pitch 72 bf16.
__device__ __forceinline__ void tdm_load_tile_2d(const void* gptr, unsigned lds_off,
                                                 unsigned tensor_d0, unsigned tensor_d1,
                                                 unsigned tile_d0, unsigned tile_d1,
                                                 unsigned stride0) {
  const unsigned long long ga = (unsigned long long)(uintptr_t)gptr;
  v4u g0;
  g0[0] = 1u;                                          // count=1 (valid user D#)
  g0[1] = lds_off;                                     // lds_addr (bytes)
  g0[2] = (unsigned)(ga & 0xffffffffu);                // global_addr[31:0]
  g0[3] = (unsigned)((ga >> 32) & 0x01ffffffu)         // global_addr[56:32]
        | (2u << 30);                                  // type=2 ("image")
  v8i_t g1;
  g1[0] = (int)((1u << 16)                             // data_size = 2 bytes
              | (1u << 20)                             // pad_enable
              | (4u << 22)                             // pad_interval: 32 dwords
              | (3u << 25));                           // pad_amount: 4 dwords
  g1[1] = (int)((tensor_d0 & 0xffffu) << 16);          // tensor_dim0[15:0]
  g1[2] = (int)(((tensor_d0 >> 16) & 0xffffu)          // tensor_dim0[31:16]
              | ((tensor_d1 & 0xffffu) << 16));        // tensor_dim1[15:0]
  g1[3] = (int)(((tensor_d1 >> 16) & 0xffffu)          // tensor_dim1[31:16]
              | ((tile_d0 & 0xffffu) << 16));          // tile_dim0
  g1[4] = (int)(tile_d1 & 0xffffu);                    // tile_dim1 (tile_dim2=0)
  g1[5] = (int)stride0;                                // tensor_dim0_stride[31:0]
  g1[6] = 0;                                           // stride0 hi / dim1_stride lo
  g1[7] = 0;
  const v4i_t z4 = {0, 0, 0, 0};
#if __clang_major__ >= 23
  const v8i_t z8 = {0, 0, 0, 0, 0, 0, 0, 0};
  __builtin_amdgcn_tensor_load_to_lds(g0, g1, z4, z4, z8, 0);
#else
  __builtin_amdgcn_tensor_load_to_lds(g0, g1, z4, z4, 0);
#endif
}
#endif

// ---------------------------------------------------------------------------
// Kernel 1: qkv[b, o, n] = sum_c w_qkv[o, c] * x[b, c, n]   (o<768), bf16 out
// grid (N/64, 768/64, B), block 128 (4 waves, each owns a 16-row strip)
// ---------------------------------------------------------------------------
__global__ void qkv_gemm_kernel(const float* __restrict__ x,
                                const float* __restrict__ w_qkv,
                                __bf16* __restrict__ qkv) {
  const int n0 = blockIdx.x * 64;
  const int o0 = blockIdx.y * 64;
  const int b  = blockIdx.z;
  const int t = threadIdx.x;
  const int wave = t >> 5, lane = t & 31;
  const int half = lane >> 4, lm = lane & 15;

  __shared__ __bf16 wl[64][40];  // [o][c] tile
  __shared__ __bf16 xl[32][72];  // [c][n] tile

  v8f acc[4] = {};
  const float* xb = x + (size_t)b * C_IN * N_TOK;

  for (int c0 = 0; c0 < C_IN; c0 += 32) {
    __syncthreads();
#pragma unroll
    for (int k = 0; k < 4; ++k) {               // 64x32 w tile, float4 chunks
      const int idx = t + k * 128;
      const int oo = idx >> 3, q = (idx & 7) * 4;
      cvt_store4(&wl[oo][q], &w_qkv[(o0 + oo) * C_IN + c0 + q]);
    }
#pragma unroll
    for (int k = 0; k < 4; ++k) {               // 32x64 x tile, float4 chunks
      const int idx = t + k * 128;
      const int cc = idx >> 4, q = (idx & 15) * 4;
      cvt_store4(&xl[cc][q], &xb[(size_t)(c0 + cc) * N_TOK + n0 + q]);
    }
    __syncthreads();
    const v16bf a = load_A(&wl[0][0], 40, wave * 16, 0);
#pragma unroll
    for (int nc = 0; nc < 4; ++nc)
      acc[nc] = wmma_bf16(a, load_B(&xl[0][0], 72, 0, nc * 16), acc[nc]);
  }

  __bf16* qb = qkv + (size_t)b * 768 * N_TOK;
#pragma unroll
  for (int nc = 0; nc < 4; ++nc)
#pragma unroll
    for (int r = 0; r < 8; ++r) {
      const int o = o0 + wave * 16 + r + 8 * half;
      const int n = n0 + nc * 16 + lm;
      qb[(size_t)o * N_TOK + n] = (__bf16)acc[nc][r];
    }
}

// ---------------------------------------------------------------------------
// Kernel 2: flash attention per (b, h, 64-query tile). block 128 = 4 waves.
// ---------------------------------------------------------------------------
__global__ void attn_kernel(const __bf16* __restrict__ qkv,
                            __bf16* __restrict__ ao) {
  const int i0 = blockIdx.x * 64;
  const int h  = blockIdx.y;
  const int b  = blockIdx.z;
  const int t = threadIdx.x;
  const int wave = t >> 5, lane = t & 31;
  const int half = lane >> 4, lm = lane & 15;

  __shared__ __bf16 qs[64][72];      // [i][d]  (Q transposed on stage-in)
  __shared__ __bf16 ks[64][72];      // [d][j]  (TDM-loaded, padded pitch 72)
  __shared__ __bf16 vs[64][72];      // [j][d]  (V transposed on stage-in)
  __shared__ __bf16 ps[4][16][72];   // per-wave P tile [i][j]

  const size_t base = (size_t)b * 768 * N_TOK;
  const __bf16* qg = qkv + base + (size_t)(h * HDIM) * N_TOK;
  const __bf16* kg = qkv + base + (size_t)(256 + h * HDIM) * N_TOK;
  const __bf16* vg = qkv + base + (size_t)(512 + h * HDIM) * N_TOK;

  // Stage Q tile (64 x 64), transposing [d][i] -> [i][d]; b128 global reads.
#pragma unroll
  for (int k = 0; k < 4; ++k) {
    const int idx = t + k * 128;
    const int d = idx >> 3, ic = (idx & 7) * 8;
    const v8bf v = *(const v8bf*)&qg[(size_t)d * N_TOK + i0 + ic];
#pragma unroll
    for (int e = 0; e < 8; ++e) qs[ic + e][d] = v[e];
  }
  __syncthreads();
  const v16bf aq0 = load_A(&qs[0][0], 72, wave * 16, 0);
  const v16bf aq1 = load_A(&qs[0][0], 72, wave * 16, 32);

  float mrow[8], lrow[8];
#pragma unroll
  for (int r = 0; r < 8; ++r) { mrow[r] = -1e30f; lrow[r] = 0.f; }
  v8f O[4] = {};

  for (int jt = 0; jt < N_TOK / 64; ++jt) {
    const int j0 = jt * 64;
    __syncthreads();

    // --- K tile: TDM if available, else async copy, else manual b128 copy ---
#if defined(HAVE_TDM)
    if (wave == 0) {
      tdm_load_tile_2d(kg + j0, (unsigned)(uintptr_t)&ks[0][0],
                       N_TOK, HDIM, 64, 64, N_TOK);
    }
#elif defined(HAVE_ASYNC)
#pragma unroll
    for (int k = 0; k < 4; ++k) {
      const int idx = t + k * 128;
      const int d = idx >> 3, jc = (idx & 7) * 8;
      __builtin_amdgcn_global_load_async_to_lds_b128(
          to_as1(&kg[(size_t)d * N_TOK + j0 + jc]), to_as3(&ks[d][jc]), 0, 0);
    }
#else
#pragma unroll
    for (int k = 0; k < 4; ++k) {
      const int idx = t + k * 128;
      const int d = idx >> 3, jc = (idx & 7) * 8;
      *(v8bf*)&ks[d][jc] = *(const v8bf*)&kg[(size_t)d * N_TOK + j0 + jc];
    }
#endif

    // --- V tile: b128 reads, transpose [d][j] -> [j][d] into LDS ---
#pragma unroll
    for (int k = 0; k < 4; ++k) {
      const int idx = t + k * 128;
      const int d = idx >> 3, jc = (idx & 7) * 8;
      const v8bf v = *(const v8bf*)&vg[(size_t)d * N_TOK + j0 + jc];
#pragma unroll
      for (int e = 0; e < 8; ++e) vs[jc + e][d] = v[e];
    }

    if (jt + 1 < N_TOK / 64) {  // hint next K tile toward the caches
      __builtin_prefetch(&kg[(size_t)(t & 63) * N_TOK + j0 + 64], 0, 1);
    }

#if defined(HAVE_TDM)
    if (wave == 0) __builtin_amdgcn_s_wait_tensorcnt(0);
#elif defined(HAVE_ASYNC)
    asm volatile("s_wait_asynccnt 0x0" ::: "memory");
#endif
    __syncthreads();

    // S = (Q^T K) * scale : four 16x16 f32 tiles per wave
    v8f S[4];
#pragma unroll
    for (int nc = 0; nc < 4; ++nc) {
      v8f s = {};
      s = wmma_bf16(aq0, load_B(&ks[0][0], 72, 0,  nc * 16), s);
      s = wmma_bf16(aq1, load_B(&ks[0][0], 72, 32, nc * 16), s);
      S[nc] = s * SCALE;
    }

    // Online softmax: slot r covers row (r + 8*half) across one 16-lane half.
#pragma unroll
    for (int r = 0; r < 8; ++r) {
      float mx = -1e30f;
#pragma unroll
      for (int nc = 0; nc < 4; ++nc) mx = fmaxf(mx, S[nc][r]);
      for (int off = 1; off < 16; off <<= 1) mx = fmaxf(mx, __shfl_xor(mx, off, 16));
      const float mnew  = fmaxf(mrow[r], mx);
      const float alpha = __expf(mrow[r] - mnew);
      float rs = 0.f;
#pragma unroll
      for (int nc = 0; nc < 4; ++nc) {
        const float p = __expf(S[nc][r] - mnew);
        rs += p;
        ps[wave][r + 8 * half][nc * 16 + lm] = (__bf16)p;
      }
      for (int off = 1; off < 16; off <<= 1) rs += __shfl_xor(rs, off, 16);
      lrow[r] = lrow[r] * alpha + rs;
      mrow[r] = mnew;
#pragma unroll
      for (int dc = 0; dc < 4; ++dc) O[dc][r] *= alpha;
    }

    // P was just written to per-wave LDS; make the RAW explicit.
    asm volatile("s_wait_dscnt 0x0" ::: "memory");

    const v16bf ap0 = load_A(&ps[wave][0][0], 72, 0, 0);
    const v16bf ap1 = load_A(&ps[wave][0][0], 72, 0, 32);
#pragma unroll
    for (int dc = 0; dc < 4; ++dc) {
      O[dc] = wmma_bf16(ap0, load_B(&vs[0][0], 72, 0,  dc * 16), O[dc]);
      O[dc] = wmma_bf16(ap1, load_B(&vs[0][0], 72, 32, dc * 16), O[dc]);
    }
  }

  // Epilogue: O / l -> ao[b, h*64+d, i] (bf16)
  __bf16* aob = ao + (size_t)b * 256 * N_TOK;
#pragma unroll
  for (int r = 0; r < 8; ++r) {
    const float inv = 1.f / lrow[r];
    const int i = i0 + wave * 16 + r + 8 * half;
#pragma unroll
    for (int dc = 0; dc < 4; ++dc) {
      const int d = dc * 16 + lm;
      aob[(size_t)(h * HDIM + d) * N_TOK + i] = (__bf16)(O[dc][r] * inv);
    }
  }
}

// ---------------------------------------------------------------------------
// Kernel 3: out[b, o, n] = sum_c w_proj[o, c] * ao[b, c, n] + x[b, o, n]
// grid (N/64, 512/64, B), block 128
// ---------------------------------------------------------------------------
__global__ void proj_kernel(const __bf16* __restrict__ ao,
                            const float* __restrict__ w_proj,
                            const float* __restrict__ x,
                            float* __restrict__ out) {
  const int n0 = blockIdx.x * 64;
  const int o0 = blockIdx.y * 64;
  const int b  = blockIdx.z;
  const int t = threadIdx.x;
  const int wave = t >> 5, lane = t & 31;
  const int half = lane >> 4, lm = lane & 15;

  __shared__ __bf16 wl[64][40];
  __shared__ __bf16 al[32][72];

  v8f acc[4] = {};
  const __bf16* aob = ao + (size_t)b * 256 * N_TOK;

  for (int c0 = 0; c0 < 256; c0 += 32) {
    __syncthreads();
#pragma unroll
    for (int k = 0; k < 4; ++k) {               // 64x32 w tile, float4 chunks
      const int idx = t + k * 128;
      const int oo = idx >> 3, q = (idx & 7) * 4;
      cvt_store4(&wl[oo][q], &w_proj[(o0 + oo) * 256 + c0 + q]);
    }
    // 32x64 bf16 B tile: async global->LDS when available.
#if defined(HAVE_ASYNC)
#pragma unroll
    for (int k = 0; k < 2; ++k) {
      const int idx = t + k * 128;
      const int cc = idx >> 3, q = (idx & 7) * 8;
      __builtin_amdgcn_global_load_async_to_lds_b128(
          to_as1(&aob[(size_t)(c0 + cc) * N_TOK + n0 + q]), to_as3(&al[cc][q]), 0, 0);
    }
    asm volatile("s_wait_asynccnt 0x0" ::: "memory");
#else
#pragma unroll
    for (int k = 0; k < 2; ++k) {
      const int idx = t + k * 128;
      const int cc = idx >> 3, q = (idx & 7) * 8;
      *(v8bf*)&al[cc][q] = *(const v8bf*)&aob[(size_t)(c0 + cc) * N_TOK + n0 + q];
    }
#endif
    __syncthreads();
    const v16bf a = load_A(&wl[0][0], 40, wave * 16, 0);
#pragma unroll
    for (int nc = 0; nc < 4; ++nc)
      acc[nc] = wmma_bf16(a, load_B(&al[0][0], 72, 0, nc * 16), acc[nc]);
  }

  const float* xb = x   + (size_t)b * C_IN * N_TOK;
  float*       ob = out + (size_t)b * C_IN * N_TOK;
#pragma unroll
  for (int nc = 0; nc < 4; ++nc)
#pragma unroll
    for (int r = 0; r < 8; ++r) {
      const int o = o0 + wave * 16 + r + 8 * half;
      const int n = n0 + nc * 16 + lm;
      const size_t idx = (size_t)o * N_TOK + n;
      ob[idx] = acc[nc][r] + xb[idx];
    }
}

// ---------------------------------------------------------------------------
extern "C" void kernel_launch(void* const* d_in, const int* in_sizes, int n_in,
                              void* d_out, int out_size, void* d_ws, size_t ws_size,
                              hipStream_t stream) {
  const float* x      = (const float*)d_in[0];
  const float* w_qkv  = (const float*)d_in[1];
  const float* w_proj = (const float*)d_in[2];
  float* out = (float*)d_out;

  // Workspace: qkv bf16 [2][768][4096] then ao bf16 [2][256][4096]
  __bf16* qkv = (__bf16*)d_ws;
  __bf16* ao  = (__bf16*)((char*)d_ws + (size_t)2 * 768 * N_TOK * sizeof(__bf16));

  const dim3 blk(128);
  qkv_gemm_kernel<<<dim3(N_TOK / 64, 768 / 64, 2), blk, 0, stream>>>(x, w_qkv, qkv);
  attn_kernel<<<dim3(N_TOK / 64, HEADS, 2), blk, 0, stream>>>(qkv, ao);
  proj_kernel<<<dim3(N_TOK / 64, C_IN / 64, 2), blk, 0, stream>>>(ao, w_proj, x, out);
}
